// TransformerLayerShard_78005196030418
// MI455X (gfx1250) — matrix operand order repaired and checked by reading, not compiled, not run on hardware
//
#include <hip/hip_runtime.h>
#include <math.h>

// ---------- CDNA5 WMMA types ----------
typedef __attribute__((ext_vector_type(16))) __bf16 v16bf;
typedef __attribute__((ext_vector_type(8)))  __bf16 v8bf;
typedef __attribute__((ext_vector_type(4)))  __bf16 v4bf;
typedef __attribute__((ext_vector_type(8)))  float  v8f;

#define D_MODEL 4096
#define N_HEADS 16
#define D_HEAD  256
#define ROTARY  64
#define D_FF    16384
#define SEQ_T   2048
#define BATCH   2
#define NTOK    (BATCH * SEQ_T)

#define BM 128
#define BN 128
#define BK 32

#define QT 64
#define KT 32
#define NEG_INF_F (-1.0e10f)

// per-lane fragment row: 16 bf16 used, padded to 20 (40B stride -> conflict-free b64s)
#define FS 20

union bf4u { __bf16 h[4]; unsigned long long u64; };

__device__ __forceinline__ float gelu_f(float x) {
    float x3 = x * x * x;
    return 0.5f * x * (1.0f + tanhf(0.7978845608028654f * (x + 0.044715f * x3)));
}

// Load one 16-element bf16 fragment for this lane: ds_load_b64 x4 (pairs merge
// into ds_load_2addr_b64), zero repacking.
__device__ __forceinline__ v16bf load_frag(const __bf16* p) {
    v4bf a = *(const v4bf*)(p);
    v4bf b = *(const v4bf*)(p + 4);
    v4bf c = *(const v4bf*)(p + 8);
    v4bf d = *(const v4bf*)(p + 12);
    v8bf ab = __builtin_shufflevector(a, b, 0, 1, 2, 3, 4, 5, 6, 7);
    v8bf cd = __builtin_shufflevector(c, d, 0, 1, 2, 3, 4, 5, 6, 7);
    return __builtin_shufflevector(ab, cd, 0, 1, 2, 3, 4, 5, 6, 7,
                                   8, 9, 10, 11, 12, 13, 14, 15);
}

__device__ __forceinline__ void pack4(bf4u& t, float a, float b, float c, float d) {
    t.h[0] = (__bf16)a; t.h[1] = (__bf16)b; t.h[2] = (__bf16)c; t.h[3] = (__bf16)d;
}

// ======================= LayerNorm =======================
__global__ __launch_bounds__(256) void ln_kernel(const float* __restrict__ x,
                                                 const float* __restrict__ g,
                                                 const float* __restrict__ b,
                                                 float* __restrict__ y)
{
    __shared__ float red[256];
    const int tid = threadIdx.x;
    const size_t row = blockIdx.x;
    const float* xr = x + row * D_MODEL;

    float loc[16];
    float s = 0.f;
#pragma unroll
    for (int i = 0; i < 16; ++i) { loc[i] = xr[tid + 256 * i]; s += loc[i]; }
    red[tid] = s; __syncthreads();
    for (int off = 128; off > 0; off >>= 1) {
        if (tid < off) red[tid] += red[tid + off];
        __syncthreads();
    }
    const float mean = red[0] * (1.0f / D_MODEL);
    __syncthreads();

    float vv = 0.f;
#pragma unroll
    for (int i = 0; i < 16; ++i) { float d = loc[i] - mean; vv += d * d; }
    red[tid] = vv; __syncthreads();
    for (int off = 128; off > 0; off >>= 1) {
        if (tid < off) red[tid] += red[tid + off];
        __syncthreads();
    }
    const float rinv = rsqrtf(red[0] * (1.0f / D_MODEL) + 1e-5f);

    float* yr = y + row * D_MODEL;
#pragma unroll
    for (int i = 0; i < 16; ++i) {
        int c = tid + 256 * i;
        yr[c] = (loc[i] - mean) * rinv * g[c] + b[c];
    }
}

// ======================= bf16 WMMA GEMM (double-buffered) =======================
// C[M,N] = A[M,K] @ B[K,N]  (+bias) (+gelu) (+residual)
// EPI: 0 = plain, 1 = +residual, 2 = +bias,gelu, 3 = +bias,+residual
// LDS tiles stored PRE-SWIZZLED into WMMA fragment layout; software-pipelined
// so tile t+1's global loads overlap tile t's WMMAs.
template <int EPI>
__global__ __launch_bounds__(256) void gemm_kernel(
    const float* __restrict__ A, const float* __restrict__ B,
    float* __restrict__ C, const float* __restrict__ bias,
    const float* __restrict__ res, int M, int N, int K)
{
    __shared__ __bf16 sAf[2][8][32][FS];   // [buf][m-block][lane][elem]
    __shared__ __bf16 sBf[2][8][32][FS];   // [buf][n-block][lane][elem]

    const int tid  = threadIdx.x;
    const int lane = tid & 31;
    const int wave = tid >> 5;
    const int lrow = lane & 15;
    const int lhi  = lane >> 4;
    const int wm = wave >> 1;   // 0..3 : 32-row block
    const int wn = wave & 1;    // 0..1 : 64-col block
    const int m0 = blockIdx.y * BM;
    const int n0 = blockIdx.x * BN;

    // staging indices (constant across k-steps)
    const int a_m[4]  = { (tid + 0) >> 3, (tid + 256) >> 3, (tid + 512) >> 3, (tid + 768) >> 3 };
    const int a_kk = (tid & 7) * 4;                       // 0..28
    const int b_n  = tid & 127;
    const int b_kk0 = (tid >> 7) * 4;                     // 0 or 4 (then +8 per it)

    v8f acc[2][4];
#pragma unroll
    for (int i = 0; i < 2; ++i)
#pragma unroll
        for (int j = 0; j < 4; ++j)
#pragma unroll
            for (int r = 0; r < 8; ++r) acc[i][j][r] = 0.f;

    float4 ra[4];
    float  rb[4][4];

    // ---- global load of tile kt into registers ----
    auto gload = [&](int k0) {
#pragma unroll
        for (int it = 0; it < 4; ++it)
            ra[it] = *(const float4*)&A[(size_t)(m0 + a_m[it]) * K + k0 + a_kk];
#pragma unroll
        for (int it = 0; it < 4; ++it) {
            int kk = b_kk0 + 8 * it;
            const float* bp = &B[(size_t)(k0 + kk) * N + n0 + b_n];
            rb[it][0] = bp[0];
            rb[it][1] = bp[(size_t)N];
            rb[it][2] = bp[2 * (size_t)N];
            rb[it][3] = bp[3 * (size_t)N];
        }
    };
    // ---- convert + store registers into fragment-layout LDS buffer ----
    auto lstore = [&](int buf) {
#pragma unroll
        for (int it = 0; it < 4; ++it) {
            int m = a_m[it];
            int ln = (m & 15) + 16 * ((a_kk >> 3) & 1);
            int e  = (a_kk & 7) + 8 * (a_kk >> 4);
            bf4u t; pack4(t, ra[it].x, ra[it].y, ra[it].z, ra[it].w);
            *(unsigned long long*)&sAf[buf][m >> 4][ln][e] = t.u64;
        }
#pragma unroll
        for (int it = 0; it < 4; ++it) {
            int kk = b_kk0 + 8 * it;
            int ln = (b_n & 15) + 16 * ((kk >> 3) & 1);
            int e  = (kk & 7) + 8 * (kk >> 4);
            bf4u t; pack4(t, rb[it][0], rb[it][1], rb[it][2], rb[it][3]);
            *(unsigned long long*)&sBf[buf][b_n >> 4][ln][e] = t.u64;
        }
    };

    const int nk = K / BK;
    gload(0);
    lstore(0);
    __syncthreads();

    for (int kt = 0; kt < nk; ++kt) {
        const int cur = kt & 1;
        if (kt + 1 < nk) gload((kt + 1) * BK);   // overlap with WMMAs below

        v16bf af[2];
#pragma unroll
        for (int t = 0; t < 2; ++t)
            af[t] = load_frag(&sAf[cur][2 * wm + t][lane][0]);
        v16bf bfv[4];
#pragma unroll
        for (int t = 0; t < 4; ++t)
            bfv[t] = load_frag(&sBf[cur][4 * wn + t][lane][0]);

#pragma unroll
        for (int i = 0; i < 2; ++i)
#pragma unroll
            for (int j = 0; j < 4; ++j)
                acc[i][j] = __builtin_amdgcn_wmma_f32_16x16x32_bf16(
                    false, af[i], false, bfv[j], (short)0, acc[i][j], false, false);

        if (kt + 1 < nk) lstore(cur ^ 1);
        __syncthreads();
    }

    // epilogue: C/D layout -> lane 0-15: M=r, lane 16-31: M=r+8; N=lane&15
#pragma unroll
    for (int i = 0; i < 2; ++i)
#pragma unroll
        for (int j = 0; j < 4; ++j)
#pragma unroll
            for (int r = 0; r < 8; ++r) {
                int row = m0 + 32 * wm + 16 * i + r + lhi * 8;
                int col = n0 + 64 * wn + 16 * j + lrow;
                float val = acc[i][j][r];
                if (EPI == 2 || EPI == 3) val += bias[col];
                if (EPI == 2) val = gelu_f(val);
                size_t idx = (size_t)row * N + col;
                if (EPI == 1 || EPI == 3) val += res[idx];
                C[idx] = val;
            }
}

// ======================= RoPE (in place, q and k) =======================
__global__ __launch_bounds__(256) void rope_kernel(float* __restrict__ q,
                                                   float* __restrict__ k)
{
    int idx = blockIdx.x * 256 + threadIdx.x;   // NTOK * H * (ROTARY/2)
    int d = idx & 31;
    int h = (idx >> 5) & 15;
    int m = idx >> 9;
    int pos = m & (SEQ_T - 1);
    float inv_freq = __expf(-logf(10000.f) * ((float)d / 32.0f));
    float fr = (float)pos * inv_freq;
    float c = cosf(fr), s = sinf(fr);
    size_t base = (size_t)m * (N_HEADS * D_HEAD) + (size_t)h * D_HEAD + 2 * d;
    float q1 = q[base], q2 = q[base + 1];
    q[base]     = q1 * c - q2 * s;
    q[base + 1] = q2 * c + q1 * s;
    float k1 = k[base], k2 = k[base + 1];
    k[base]     = k1 * c - k2 * s;
    k[base + 1] = k2 * c + k1 * s;
}

// ======================= Flash attention (causal) =======================
// One block = (b, h, 64-query tile). 8 waves: wave>>1 -> query row block,
// wave&1 -> key half. K/V staged in fragment layout; softmax emits P
// directly in packed A-fragment layout.
__global__ __launch_bounds__(256) void flash_kernel(
    const float* __restrict__ q, const float* __restrict__ k,
    const float* __restrict__ v, float* __restrict__ ctx)
{
    __shared__ __bf16 sKf[2][8][32][FS];   // [key-block][d-chunk][lane][elem]
    __shared__ __bf16 sVf[16][32][FS];     // [col-block][lane][elem]
    __shared__ __bf16 sPf[4][32][FS];      // [q-row-block][lane][elem]
    __shared__ float  sS[QT][KT + 1];
    __shared__ float  sM[QT], sL[QT], sScale[QT];

    const int tid  = threadIdx.x;
    const int lane = tid & 31;
    const int wave = tid >> 5;
    const int lrow = lane & 15;
    const int lhi  = lane >> 4;
    const int mq = wave >> 1;   // 0..3
    const int nq = wave & 1;    // 0..1
    const int b  = blockIdx.z, h = blockIdx.y;
    const int q0 = blockIdx.x * QT;
    const size_t hoff = (size_t)h * D_HEAD;
    const float scale = 1.0f / 16.0f;           // 1/sqrt(256)

    // Q fragments in registers (one-time gather per block)
    v16bf qf[8];
    {
        int tok = b * SEQ_T + q0 + 16 * mq + lrow;
        const float* qp = q + (size_t)tok * (N_HEADS * D_HEAD) + hoff;
        int kh = lhi * 8;
#pragma unroll
        for (int c = 0; c < 8; ++c) {
#pragma unroll
            for (int vv = 0; vv < 8; ++vv) {
                int kb = (vv < 4 ? 2 * vv : 2 * vv + 8) + kh;
                qf[c][2 * vv]     = (__bf16)qp[32 * c + kb];
                qf[c][2 * vv + 1] = (__bf16)qp[32 * c + kb + 1];
            }
        }
    }

    if (tid < QT) { sM[tid] = -1e30f; sL[tid] = 0.f; }

    v8f o[8];
#pragma unroll
    for (int j = 0; j < 8; ++j)
#pragma unroll
        for (int r = 0; r < 8; ++r) o[j][r] = 0.f;
    __syncthreads();

    for (int kb0 = 0; kb0 < q0 + QT; kb0 += KT) {
        // ---- stage K tile (32 keys x 256 d) into S-B-fragment layout ----
#pragma unroll
        for (int it = 0; it < 8; ++it) {
            int i = tid + 256 * it;              // 0..2047
            int key = i >> 6;                    // 0..31
            int d   = (i & 63) * 4;              // 0..252
            int tok = b * SEQ_T + kb0 + key;
            const float4 f = *(const float4*)&k[(size_t)tok * (N_HEADS * D_HEAD) + hoff + d];
            int ln = (key & 15) + 16 * ((d >> 3) & 1);
            int e  = (d & 7) + 8 * ((d >> 4) & 1);
            bf4u t; pack4(t, f.x, f.y, f.z, f.w);
            *(unsigned long long*)&sKf[key >> 4][d >> 5][ln][e] = t.u64;
        }
        // ---- stage V tile into PV-B-fragment layout ----
#pragma unroll
        for (int it = 0; it < 8; ++it) {
            int i = tid + 256 * it;              // 0..2047
            int col = i & 255;
            int key = (i >> 8) * 4;              // 0,4,...,28
            size_t base = (size_t)(b * SEQ_T + kb0 + key) * (N_HEADS * D_HEAD) + hoff + col;
            const size_t rstride = (size_t)(N_HEADS * D_HEAD);
            bf4u t; pack4(t, v[base], v[base + rstride], v[base + 2 * rstride], v[base + 3 * rstride]);
            int ln = (col & 15) + 16 * ((key >> 3) & 1);
            int e  = (key & 7) + 8 * (key >> 4);
            *(unsigned long long*)&sVf[col >> 4][ln][e] = t.u64;
        }
        __syncthreads();

        // S tile (16x16) for this wave: contract over Dh=256 in 8 WMMA
        v8f sacc;
#pragma unroll
        for (int r = 0; r < 8; ++r) sacc[r] = 0.f;
#pragma unroll
        for (int c = 0; c < 8; ++c) {
            v16bf bfr = load_frag(&sKf[nq][c][lane][0]);
            sacc = __builtin_amdgcn_wmma_f32_16x16x32_bf16(
                false, qf[c], false, bfr, (short)0, sacc, false, false);
        }

        // stage S with scale + causal mask
#pragma unroll
        for (int r = 0; r < 8; ++r) {
            int srow = 16 * mq + r + lhi * 8;
            int scol = 16 * nq + lrow;
            float sv = sacc[r] * scale;
            if (kb0 + scol > q0 + srow) sv = NEG_INF_F;
            sS[srow][scol] = sv;
        }
        __syncthreads();

        // online softmax: one thread per query row; emits P packed in
        // A-fragment layout (b64 stores)
        if (tid < QT) {
            float mold = sM[tid];
            float tmax = sS[tid][0];
#pragma unroll
            for (int c = 1; c < KT; ++c) tmax = fmaxf(tmax, sS[tid][c]);
            float mnew = fmaxf(mold, tmax);
            float sc = __expf(mold - mnew);
            float lsum = 0.f;
            int lanelo = tid & 15;
#pragma unroll
            for (int kq = 0; kq < 8; ++kq) {
                int kk = 4 * kq;
                float p0 = __expf(sS[tid][kk]     - mnew);
                float p1 = __expf(sS[tid][kk + 1] - mnew);
                float p2 = __expf(sS[tid][kk + 2] - mnew);
                float p3 = __expf(sS[tid][kk + 3] - mnew);
                lsum += p0 + p1 + p2 + p3;
                bf4u t; pack4(t, p0, p1, p2, p3);
                int ln = lanelo + 16 * ((kk >> 3) & 1);
                int e  = (kk & 7) + 8 * (kk >> 4);
                *(unsigned long long*)&sPf[tid >> 4][ln][e] = t.u64;
            }
            sM[tid] = mnew;
            sL[tid] = sL[tid] * sc + lsum;
            sScale[tid] = sc;
        }
        __syncthreads();

        // rescale O and accumulate P @ V
#pragma unroll
        for (int r = 0; r < 8; ++r) {
            float f = sScale[16 * mq + r + lhi * 8];
#pragma unroll
            for (int j = 0; j < 8; ++j) o[j][r] *= f;
        }
        v16bf pf = load_frag(&sPf[mq][lane][0]);
#pragma unroll
        for (int j = 0; j < 8; ++j) {
            v16bf bfr = load_frag(&sVf[8 * nq + j][lane][0]);
            o[j] = __builtin_amdgcn_wmma_f32_16x16x32_bf16(
                false, pf, false, bfr, (short)0, o[j], false, false);
        }
        __syncthreads();
    }

    // normalize and write ctx
#pragma unroll
    for (int r = 0; r < 8; ++r) {
        int lr = 16 * mq + r + lhi * 8;
        float invl = 1.0f / sL[lr];
        int tok = b * SEQ_T + q0 + lr;
#pragma unroll
        for (int j = 0; j < 8; ++j) {
            int col = 128 * nq + 16 * j + lrow;
            ctx[(size_t)tok * (N_HEADS * D_HEAD) + hoff + col] = o[j][r] * invl;
        }
    }
}

// ======================= launch =======================
extern "C" void kernel_launch(void* const* d_in, const int* in_sizes, int n_in,
                              void* d_out, int out_size, void* d_ws, size_t ws_size,
                              hipStream_t stream)
{
    (void)in_sizes; (void)n_in; (void)out_size; (void)ws_size;

    const float* x     = (const float*)d_in[0];
    const float* lng   = (const float*)d_in[1];
    const float* lnb   = (const float*)d_in[2];
    const float* wq    = (const float*)d_in[3];
    const float* wk    = (const float*)d_in[4];
    const float* wv    = (const float*)d_in[5];
    const float* wo    = (const float*)d_in[6];
    const float* w_in  = (const float*)d_in[7];
    const float* b_in  = (const float*)d_in[8];
    const float* w_out = (const float*)d_in[9];
    const float* b_out = (const float*)d_in[10];
    float* out = (float*)d_out;

    float* ws = (float*)d_ws;
    const size_t TOKD = (size_t)NTOK * D_MODEL;
    float* xn  = ws;
    float* qb  = ws + TOKD;
    float* kb  = ws + 2 * TOKD;
    float* vb  = ws + 3 * TOKD;
    float* hb  = ws + 4 * TOKD;     // NTOK * D_FF
    float* ctx = xn;                 // reuse xn after QKV projections
    float* xn2 = kb;                 // reuse k after attention

    // 1) LN
    ln_kernel<<<NTOK, 256, 0, stream>>>(x, lng, lnb, xn);

    // 2) QKV projections (bf16 WMMA GEMMs)
    dim3 g_sq(D_MODEL / BN, NTOK / BM);   // 32 x 32
    gemm_kernel<0><<<g_sq, 256, 0, stream>>>(xn, wq, qb, nullptr, nullptr, NTOK, D_MODEL, D_MODEL);
    gemm_kernel<0><<<g_sq, 256, 0, stream>>>(xn, wk, kb, nullptr, nullptr, NTOK, D_MODEL, D_MODEL);
    gemm_kernel<0><<<g_sq, 256, 0, stream>>>(xn, wv, vb, nullptr, nullptr, NTOK, D_MODEL, D_MODEL);

    // 3) RoPE
    rope_kernel<<<(NTOK * N_HEADS * (ROTARY / 2)) / 256, 256, 0, stream>>>(qb, kb);

    // 4) Flash attention
    dim3 g_fa(SEQ_T / QT, N_HEADS, BATCH);
    flash_kernel<<<g_fa, 256, 0, stream>>>(qb, kb, vb, ctx);

    // 5) attn_out = ctx @ wo + x  -> out
    gemm_kernel<1><<<g_sq, 256, 0, stream>>>(ctx, wo, out, nullptr, x, NTOK, D_MODEL, D_MODEL);

    // 6) LN2
    ln_kernel<<<NTOK, 256, 0, stream>>>(out, lng, lnb, xn2);

    // 7) h = gelu(xn2 @ w_in + b_in)
    dim3 g_in(D_FF / BN, NTOK / BM);      // 128 x 32
    gemm_kernel<2><<<g_in, 256, 0, stream>>>(xn2, w_in, hb, b_in, nullptr, NTOK, D_FF, D_MODEL);

    // 8) out += h @ w_out + b_out
    gemm_kernel<3><<<g_sq, 256, 0, stream>>>(hb, w_out, out, b_out, out, NTOK, D_MODEL, D_FF);
}
